// Fp8Linear_27161373180238
// MI455X (gfx1250) — 8-bit matrix operand order repaired and checked by reading, compiled, not run on hardware
//
#include <hip/hip_runtime.h>

typedef __attribute__((ext_vector_type(16))) int   v16i;
typedef __attribute__((ext_vector_type(8)))  float v8f;

#define FP8_MAX 448.0f
#define QEPS 1e-12f

#define BM 128
#define BN 128
#define BK 128

__device__ __forceinline__ void async_copy_b128(unsigned lds_off,
                                                const void* gptr) {
  asm volatile("global_load_async_to_lds_b128 %0, %1, off"
               :
               : "v"(lds_off), "v"((unsigned long long)(size_t)gptr)
               : "memory");
}
__device__ __forceinline__ void wait_async0() {
  asm volatile("s_wait_asynccnt 0" ::: "memory");
}
__device__ __forceinline__ unsigned lds_offset(const void* p) {
  return (unsigned)(size_t)p;
}

__global__ __launch_bounds__(256) void fp8_quant_rows(
    const float* __restrict__ x, unsigned char* __restrict__ q,
    float* __restrict__ recip, int K) {
  const int row = blockIdx.x;
  const float* xr = x + (size_t)row * K;

  float amax = 0.0f;
  for (int k = threadIdx.x; k < K; k += blockDim.x)
    amax = fmaxf(amax, fabsf(xr[k]));

  __shared__ float red[256];
  red[threadIdx.x] = amax;
  __syncthreads();
#pragma unroll
  for (int s = 128; s > 0; s >>= 1) {
    if (threadIdx.x < (unsigned)s)
      red[threadIdx.x] = fmaxf(red[threadIdx.x], red[threadIdx.x + s]);
    __syncthreads();
  }
  amax = fmaxf(red[0], QEPS);
  const float scale = FP8_MAX / amax;
  if (threadIdx.x == 0) recip[row] = amax / FP8_MAX;

  unsigned int* qo = (unsigned int*)(q + (size_t)row * K);
  const float4* x4 = (const float4*)xr;
  for (int i = threadIdx.x; i < (K >> 2); i += blockDim.x) {
    float4 v = x4[i];
    float a = fminf(fmaxf(v.x * scale, -FP8_MAX), FP8_MAX);
    float b = fminf(fmaxf(v.y * scale, -FP8_MAX), FP8_MAX);
    float c = fminf(fmaxf(v.z * scale, -FP8_MAX), FP8_MAX);
    float d = fminf(fmaxf(v.w * scale, -FP8_MAX), FP8_MAX);
    int packed = 0;
    packed = __builtin_amdgcn_cvt_pk_fp8_f32(a, b, packed, false);
    packed = __builtin_amdgcn_cvt_pk_fp8_f32(c, d, packed, true);
    qo[i] = packed;
  }
}

__global__ __launch_bounds__(256) void fp8_gemm_wmma(
    const unsigned char* __restrict__ qA,
    const unsigned char* __restrict__ qB,
    const float* __restrict__ xs,
    const float* __restrict__ ws,
    const float* __restrict__ bias,
    float* __restrict__ out,
    int M, int N, int K) {
  __shared__ __align__(16) unsigned char As[2][BM][BK];
  __shared__ __align__(16) unsigned char Bs[2][BN][BK];

  const int tid  = threadIdx.x;
  const int lane = tid & 31;
  const int wave = tid >> 5;
  const int h    = lane >> 4;
  const int ln16 = lane & 15;

  const int blockM = blockIdx.y * BM;
  const int blockN = blockIdx.x * BN;
  const int waveM  = (wave >> 2) * 64;
  const int waveN  = (wave & 3) * 32;

  v8f acc[4][2];
#pragma unroll
  for (int i = 0; i < 4; ++i)
#pragma unroll
    for (int j = 0; j < 2; ++j) acc[i][j] = (v8f){};

  const int NKB = K / BK;

  const int ldRow = tid >> 3;
  const int ldCol = (tid & 7) * 16;

  const unsigned char* gA = qA + (size_t)blockM * K + ldCol;
  const unsigned char* gB = qB + (size_t)blockN * K + ldCol;

#pragma unroll
  for (int p = 0; p < 4; ++p) {
    const int row = p * 32 + ldRow;
    async_copy_b128(lds_offset(&As[0][row][ldCol]), gA + (size_t)row * K);
    async_copy_b128(lds_offset(&Bs[0][row][ldCol]), gB + (size_t)row * K);
  }
  wait_async0();
  __syncthreads();

  for (int kb = 0; kb < NKB; ++kb) {
    const int cur = kb & 1;
    const int nxt = cur ^ 1;

    if (kb + 1 < NKB) {
      const size_t koff = (size_t)(kb + 1) * BK;
#pragma unroll
      for (int p = 0; p < 4; ++p) {
        const int row = p * 32 + ldRow;
        async_copy_b128(lds_offset(&As[nxt][row][ldCol]),
                        gA + (size_t)row * K + koff);
        async_copy_b128(lds_offset(&Bs[nxt][row][ldCol]),
                        gB + (size_t)row * K + koff);
      }
    }

    v16i bfrag[2];
#pragma unroll
    for (int tn = 0; tn < 2; ++tn) {
      const unsigned char* bp = &Bs[cur][waveN + tn * 16 + ln16][0];
      uint4* bv = (uint4*)&bfrag[tn];
#pragma unroll
      for (int qd = 0; qd < 4; ++qd)
        bv[qd] = *(const uint4*)(bp + qd * 32 + h * 16);
    }

#pragma unroll
    for (int tm = 0; tm < 4; ++tm) {
      v16i afrag;
      const unsigned char* ap = &As[cur][waveM + tm * 16 + ln16][0];
      uint4* av = (uint4*)&afrag;
#pragma unroll
      for (int c = 0; c < 4; ++c)
        av[c] = *(const uint4*)(ap + c * 32 + h * 16);
#pragma unroll
      for (int tn = 0; tn < 2; ++tn)
        acc[tm][tn] = __builtin_amdgcn_wmma_f32_16x16x128_fp8_fp8(
            afrag, bfrag[tn], (short)0, acc[tm][tn],
             false,  false);
    }

    wait_async0();
    __syncthreads();
  }

#pragma unroll
  for (int tn = 0; tn < 2; ++tn) {
    const int n = blockN + waveN + tn * 16 + ln16;
    const float wsn = ws[n];
    const float bn  = bias[n];
#pragma unroll
    for (int tm = 0; tm < 4; ++tm) {
      const int mBase = blockM + waveM + tm * 16 + h * 8;
#pragma unroll
      for (int v = 0; v < 8; ++v) {
        const int m = mBase + v;
        out[(size_t)m * N + n] = acc[tm][tn][v] * xs[m] * wsn + bn;
      }
    }
  }
}

extern "C" void kernel_launch(void* const* d_in, const int* in_sizes, int n_in,
                              void* d_out, int out_size, void* d_ws,
                              size_t ws_size, hipStream_t stream) {
  const float* input  = (const float*)d_in[0];
  const float* weight = (const float*)d_in[1];
  const float* bias   = (const float*)d_in[2];

  const int N = in_sizes[2];
  const int K = in_sizes[1] / N;
  const int M = in_sizes[0] / K;

  unsigned char* qA = (unsigned char*)d_ws;
  unsigned char* qW = qA + (size_t)M * K;
  float* xsc = (float*)(qW + (size_t)N * K);
  float* wsc = xsc + M;

  fp8_quant_rows<<<M, 256, 0, stream>>>(input, qA, xsc, K);
  fp8_quant_rows<<<N, 256, 0, stream>>>(weight, qW, wsc, K);

  dim3 grid(N / BN, M / BM);
  fp8_gemm_wmma<<<grid, 256, 0, stream>>>(qA, qW, xsc, wsc, bias,
                                          (float*)d_out, M, N, K);
}
